// NaverModel_33320356282901
// MI455X (gfx1250) — matrix operand order repaired and checked by reading, compiled, not run on hardware
//
#include <hip/hip_runtime.h>
#include <hip/hip_bf16.h>

// ---------------- types ----------------
typedef __attribute__((ext_vector_type(16))) __bf16 bf16x16;
typedef __attribute__((ext_vector_type(8)))  float  f32x8;
typedef __attribute__((ext_vector_type(8)))  unsigned short u16x8;
typedef __attribute__((ext_vector_type(4)))  int    i32x4;

// ---------------- problem dims ----------------
#define Bsz 256
#define Tsz 512
#define Esz 512
#define Hsz 1024
#define Csz 128
#define G4  (4*Hsz)        // 4096 gate columns
#define K1  (Esz+Hsz)      // 1536 : K of recurrent GEMM
#define K2  (Hsz+Csz)      // 1152 : K of output linear
#define DECAYF 0.9f

// ---------------- GEMM tiling ----------------
#define BM 64              // batch rows per block
#define BH 32              // h-columns per block (gates kernel)
#define BN 128             // gate/output columns per block
#define KT 32              // K per WMMA stage (bf16 16x16x32)
#define AP 48              // LDS pitch (bf16 elems), 96B = 3x32B aligned rows
#define WP 48

// ---------------- CDNA5 async global->LDS (ASYNCcnt) ----------------
#if defined(__AMDGCN__) && __has_builtin(__builtin_amdgcn_global_load_async_to_lds_b128)
#define HAVE_ASYNC_LDS 1
#else
#define HAVE_ASYNC_LDS 0
#endif

__device__ __forceinline__ void async_cp16(void* lds, const void* g) {
#if HAVE_ASYNC_LDS
  __builtin_amdgcn_global_load_async_to_lds_b128(
      (__attribute__((address_space(1))) i32x4*)(void*)g,
      (__attribute__((address_space(3))) i32x4*)lds, 0, 0);
#else
  *(u16x8*)lds = *(const u16x8*)g;
#endif
}

__device__ __forceinline__ void wait_async() {
#if HAVE_ASYNC_LDS
#if __has_builtin(__builtin_amdgcn_s_wait_asynccnt)
  __builtin_amdgcn_s_wait_asynccnt(0);
#else
  asm volatile("s_wait_asynccnt 0x0" ::: "memory");
#endif
#endif
}

__device__ __forceinline__ unsigned short f2bf(float f) {
  unsigned int u = __float_as_uint(f);
  u += 0x7FFFu + ((u >> 16) & 1u);               // round-to-nearest-even
  return (unsigned short)(u >> 16);
}

union Frag { bf16x16 f; u16x8 h[2]; };

// =====================================================================
// Fused gates GEMM (bf16 WMMA) + LSTM cell pointwise update.
// Block = 256 threads (8 waves). Tile: M=64 batch rows x 32 h-columns,
// i.e. 128 gate columns (4 gates x 32). K = E+H = 1536 in 32-wide stages.
// All tile staging is pure 16B copies -> async global->LDS when available.
// =====================================================================
__global__ void __launch_bounds__(256) lstm_step_kernel(
    const unsigned short* __restrict__ xbf,    // [B,E] bf16 (x at step t)
    const unsigned short* __restrict__ hprev,  // [B,H] bf16
    const unsigned short* __restrict__ Wih,    // [4H,E] bf16
    const unsigned short* __restrict__ Whh,    // [4H,H] bf16
    const float* __restrict__ bias,            // [4H] = b_ih + b_hh
    float* __restrict__ cstate,                // [B,H] f32
    unsigned short* __restrict__ hnext)        // [B,H] bf16
{
  __shared__ __attribute__((aligned(32))) union {
    struct { unsigned short As[BM][AP]; unsigned short Ws[BN][WP]; } tl;
    float g[BM][BN + 4];
  } sm;

  const int tid  = threadIdx.x;
  const int wave = tid >> 5;
  const int lane = tid & 31;
  const int half = lane >> 4;      // lanes 16-31 hold K 16..31 (this code's k order) and D rows 8..15
  const int l16  = lane & 15;
  const int m0   = blockIdx.x * BM;
  const int hh0  = blockIdx.y * BH;

  f32x8 acc[4] = {};

  const int arow = tid >> 2;             // 0..63  A-tile row
  const int aseg = (tid & 3) * 8;        // 8 bf16 per thread
  const int wrow = tid >> 1;             // 0..127 W-tile n_local
  const int wseg = (tid & 1) * 16;       // 2x8 bf16 per thread
  // n_local -> global gate row j: 4 strips of 32 columns across i/f/g/o
  const int j = (wrow >> 5) * Hsz + hh0 + (wrow & 31);

  for (int kk0 = 0; kk0 < K1; kk0 += KT) {
    // ---- stage A tile: [x_t | h_prev], rows m0..m0+63, k in [kk0,kk0+32) ----
    const unsigned short* pa = (kk0 < Esz)
        ? (xbf   + (size_t)(m0 + arow) * Esz + kk0 + aseg)
        : (hprev + (size_t)(m0 + arow) * Hsz + (kk0 - Esz) + aseg);
    async_cp16(&sm.tl.As[arow][aseg], pa);
    // ---- stage W tile: 128 gate rows x 32 k (row-major => B^T layout) ----
    const unsigned short* pw = (kk0 < Esz)
        ? (Wih + (size_t)j * Esz + kk0 + wseg)
        : (Whh + (size_t)j * Hsz + (kk0 - Esz) + wseg);
    async_cp16(&sm.tl.Ws[wrow][wseg],     pw);
    async_cp16(&sm.tl.Ws[wrow][wseg + 8], pw + 8);
    wait_async();
    __syncthreads();

    // B fragment (32x16): lane holds column n=l16, K = half*16 + e  (32B aligned)
    bf16x16 bfv = *(const bf16x16*)&sm.tl.Ws[wave * 16 + l16][half * 16];
    #pragma unroll
    for (int i = 0; i < 4; ++i) {
      // A fragment (16x32): lane row m=l16; K = half*16 + e
      Frag af;
      af.h[0] = *(const u16x8*)&sm.tl.As[i * 16 + l16][half * 16];
      af.h[1] = *(const u16x8*)&sm.tl.As[i * 16 + l16][half * 16 + 8];
      acc[i] = __builtin_amdgcn_wmma_f32_16x16x32_bf16(
          false, af.f, false, bfv, (short)0, acc[i], false, false);
    }
    __syncthreads();
  }

  // ---- stage gate accumulators to LDS (D layout: reg r -> M = half*8+r) ----
  #pragma unroll
  for (int i = 0; i < 4; ++i)
    #pragma unroll
    for (int r = 0; r < 8; ++r)
      sm.g[i * 16 + half * 8 + r][wave * 16 + l16] = acc[i][r];
  __syncthreads();

  // ---- pointwise LSTM cell update for the BM x BH state tile ----
  #pragma unroll
  for (int p = 0; p < (BM * BH) / 256; ++p) {
    int idx = tid + p * 256;
    int m = idx >> 5, hh = idx & 31;
    int b = m0 + m, hg = hh0 + hh;
    float gi = sm.g[m][hh]      + bias[hg];
    float gf = sm.g[m][32 + hh] + bias[Hsz + hg];
    float gg = sm.g[m][64 + hh] + bias[2 * Hsz + hg];
    float go = sm.g[m][96 + hh] + bias[3 * Hsz + hg];
    float iv = 1.f / (1.f + __expf(-gi));
    float fv = 1.f / (1.f + __expf(-gf));
    float gv = tanhf(gg);
    float ov = 1.f / (1.f + __expf(-go));
    size_t s = (size_t)b * Hsz + hg;
    float c = fv * cstate[s] + iv * gv;
    cstate[s] = c;
    hnext[s] = f2bf(ov * tanhf(c));
  }
}

// =====================================================================
// Output linear: out[:,t,:] = [h_t | cate_eff] @ W_lin^T + b_lin,
// masked: rows with t >= seq_len get exactly b_lin (linear of zeros).
// =====================================================================
__global__ void __launch_bounds__(256) out_linear_kernel(
    const unsigned short* __restrict__ hcur,    // [B,H] bf16
    const unsigned short* __restrict__ catebf,  // [B,C] bf16
    const unsigned short* __restrict__ Wlin,    // [E,H+C] bf16
    const float* __restrict__ blin,             // [E]
    const int* __restrict__ lens,               // [B]
    float* __restrict__ out,                    // [B,T,E] f32
    int t)
{
  __shared__ __attribute__((aligned(32)))
      struct { unsigned short As[BM][AP]; unsigned short Ws[BN][WP]; } sm;
  const int tid  = threadIdx.x;
  const int wave = tid >> 5;
  const int lane = tid & 31;
  const int half = lane >> 4;
  const int l16  = lane & 15;
  const int m0   = blockIdx.x * BM;
  const int n0   = blockIdx.y * BN;

  f32x8 acc[4] = {};
  const int arow = tid >> 2;
  const int aseg = (tid & 3) * 8;
  const int wrow = tid >> 1;
  const int wseg = (tid & 1) * 16;

  for (int kk0 = 0; kk0 < K2; kk0 += KT) {
    const unsigned short* pa = (kk0 < Hsz)
        ? (hcur   + (size_t)(m0 + arow) * Hsz + kk0 + aseg)
        : (catebf + (size_t)(m0 + arow) * Csz + (kk0 - Hsz) + aseg);
    async_cp16(&sm.As[arow][aseg], pa);
    const unsigned short* pw = Wlin + (size_t)(n0 + wrow) * K2 + kk0 + wseg;
    async_cp16(&sm.Ws[wrow][wseg],     pw);
    async_cp16(&sm.Ws[wrow][wseg + 8], pw + 8);
    wait_async();
    __syncthreads();

    bf16x16 bfv = *(const bf16x16*)&sm.Ws[wave * 16 + l16][half * 16];
    #pragma unroll
    for (int i = 0; i < 4; ++i) {
      Frag af;
      af.h[0] = *(const u16x8*)&sm.As[i * 16 + l16][half * 16];
      af.h[1] = *(const u16x8*)&sm.As[i * 16 + l16][half * 16 + 8];
      acc[i] = __builtin_amdgcn_wmma_f32_16x16x32_bf16(
          false, af.f, false, bfv, (short)0, acc[i], false, false);
    }
    __syncthreads();
  }

  const int n = n0 + wave * 16 + l16;
  const float bl = blin[n];
  #pragma unroll
  for (int i = 0; i < 4; ++i) {
    #pragma unroll
    for (int r = 0; r < 8; ++r) {
      int b = m0 + i * 16 + half * 8 + r;
      float v = (t < lens[b]) ? (acc[i][r] + bl) : bl;
      out[((size_t)b * Tsz + t) * Esz + n] = v;
    }
  }
}

// ---------------- small helper kernels ----------------
__global__ void xcvt_kernel(const float* __restrict__ x,
                            unsigned short* __restrict__ xbf, int t) {
  int i = blockIdx.x * blockDim.x + threadIdx.x;   // < B*E
  int b = i >> 9, e = i & (Esz - 1);
  xbf[i] = f2bf(x[((size_t)b * Tsz + t) * Esz + e]);
}

__global__ void cate_kernel(const float* __restrict__ cate, float* __restrict__ pcate,
                            unsigned short* __restrict__ catebf, int t) {
  int i = blockIdx.x * blockDim.x + threadIdx.x;   // < B*C
  int b = i >> 7, ci = i & (Csz - 1);
  float cur = cate[((size_t)b * Tsz + t) * Csz + ci];
  float eff = (t > 0) ? (DECAYF * pcate[i] + (1.f - DECAYF) * cur) : cur;
  pcate[i] = eff;
  catebf[i] = f2bf(eff);
}

__global__ void cvt_bf16_kernel(const float* __restrict__ src,
                                unsigned short* __restrict__ dst, int n) {
  int i = blockIdx.x * blockDim.x + threadIdx.x;
  if (i < n) dst[i] = f2bf(src[i]);
}

__global__ void bias_sum_kernel(const float* __restrict__ a, const float* __restrict__ b,
                                float* __restrict__ d, int n) {
  int i = blockIdx.x * blockDim.x + threadIdx.x;
  if (i < n) d[i] = a[i] + b[i];
}

__global__ void zero_f32_kernel(float* __restrict__ p, int n) {
  int i = blockIdx.x * blockDim.x + threadIdx.x;
  if (i < n) p[i] = 0.f;
}

__global__ void zero_u16_kernel(unsigned short* __restrict__ p, int n) {
  int i = blockIdx.x * blockDim.x + threadIdx.x;
  if (i < n) p[i] = 0;
}

// =====================================================================
extern "C" void kernel_launch(void* const* d_in, const int* in_sizes, int n_in,
                              void* d_out, int out_size, void* d_ws, size_t ws_size,
                              hipStream_t stream) {
  (void)in_sizes; (void)n_in; (void)out_size; (void)ws_size;
  const float* x     = (const float*)d_in[0];
  const float* cate  = (const float*)d_in[1];
  const int*   lens  = (const int*)d_in[2];
  const float* W_ih  = (const float*)d_in[3];
  const float* W_hh  = (const float*)d_in[4];
  const float* b_ih  = (const float*)d_in[5];
  const float* b_hh  = (const float*)d_in[6];
  const float* W_lin = (const float*)d_in[7];
  const float* b_lin = (const float*)d_in[8];
  float* out = (float*)d_out;

  // ---- workspace carve-up (~16.4 MB total) ----
  char* w = (char*)d_ws; size_t off = 0;
  auto alloc = [&](size_t bytes) -> void* {
    void* p = w + off; off = (off + bytes + 255) & ~(size_t)255; return p;
  };
  unsigned short* Wih_bf  = (unsigned short*)alloc((size_t)G4 * Esz * 2);
  unsigned short* Whh_bf  = (unsigned short*)alloc((size_t)G4 * Hsz * 2);
  unsigned short* Wlin_bf = (unsigned short*)alloc((size_t)Esz * K2 * 2);
  float*          bias    = (float*)alloc((size_t)G4 * 4);
  float*          cstate  = (float*)alloc((size_t)Bsz * Hsz * 4);
  unsigned short* hbuf0   = (unsigned short*)alloc((size_t)Bsz * Hsz * 2);
  unsigned short* hbuf1   = (unsigned short*)alloc((size_t)Bsz * Hsz * 2);
  float*          pcate   = (float*)alloc((size_t)Bsz * Csz * 4);
  unsigned short* catebf  = (unsigned short*)alloc((size_t)Bsz * Csz * 2);
  unsigned short* xbf     = (unsigned short*)alloc((size_t)Bsz * Esz * 2);

  // ---- one-time (per call) prep: bf16 weights, bias fold, state init ----
  int n;
  n = G4 * Esz;  cvt_bf16_kernel<<<(n + 255) / 256, 256, 0, stream>>>(W_ih,  Wih_bf,  n);
  n = G4 * Hsz;  cvt_bf16_kernel<<<(n + 255) / 256, 256, 0, stream>>>(W_hh,  Whh_bf,  n);
  n = Esz * K2;  cvt_bf16_kernel<<<(n + 255) / 256, 256, 0, stream>>>(W_lin, Wlin_bf, n);
  n = G4;        bias_sum_kernel<<<(n + 255) / 256, 256, 0, stream>>>(b_ih, b_hh, bias, n);
  n = Bsz * Hsz; zero_f32_kernel<<<(n + 255) / 256, 256, 0, stream>>>(cstate, n);
  n = Bsz * Hsz; zero_u16_kernel<<<(n + 255) / 256, 256, 0, stream>>>(hbuf0, n);

  // ---- recurrent scan over T steps ----
  const dim3 gGates(Bsz / BM, Hsz / BH);   // (4, 32)
  const dim3 gLin  (Bsz / BM, Esz / BN);   // (4, 4)
  for (int t = 0; t < Tsz; ++t) {
    unsigned short* hprev = (t & 1) ? hbuf1 : hbuf0;
    unsigned short* hnext = (t & 1) ? hbuf0 : hbuf1;
    xcvt_kernel<<<(Bsz * Esz) / 256, 256, 0, stream>>>(x, xbf, t);
    cate_kernel<<<(Bsz * Csz) / 256, 256, 0, stream>>>(cate, pcate, catebf, t);
    lstm_step_kernel<<<gGates, 256, 0, stream>>>(xbf, hprev, Wih_bf, Whh_bf,
                                                 bias, cstate, hnext);
    out_linear_kernel<<<gLin, 256, 0, stream>>>(hnext, catebf, Wlin_bf,
                                                b_lin, lens, out, t);
  }
}
